// SimpleMAMBA_74895639708063
// MI455X (gfx1250) — matrix operand order repaired and checked by reading, not compiled, not run on hardware
//
#include <hip/hip_runtime.h>
#include <hip/hip_bf16.h>

// ---------------- problem constants ----------------
#define NB     4
#define LSEQ   2048
#define NTOK   (NB * LSEQ)        // 8192 tokens
#define DIN    512
#define DM     1024
#define DI     2048               // d_inner
#define DSTATE 16
#define DTR    64                 // dt_rank
#define DBCW   96                 // dt_rank + 2*d_state
#define DBCP   128                // padded to multiple of 64
#define DOUTW  512

typedef __attribute__((ext_vector_type(16))) __bf16 bf16x16;
typedef __attribute__((ext_vector_type(8)))  __bf16 bf16x8;
typedef __attribute__((ext_vector_type(8)))  float  f32x8;

__device__ __forceinline__ unsigned short f2bf(float f) {
    unsigned u = __builtin_bit_cast(unsigned, f);
    unsigned r = u + 0x7FFFu + ((u >> 16) & 1u);   // round-to-nearest-even
    return (unsigned short)(r >> 16);
}
__device__ __forceinline__ float bf2f(unsigned short h) {
    return __builtin_bit_cast(float, (unsigned)h << 16);
}
__device__ __forceinline__ float siluf(float x) { return x / (1.f + __expf(-x)); }
__device__ __forceinline__ float softplusf(float v) {
    return v > 20.f ? v : log1pf(__expf(v));
}

union FragU { bf16x16 v; bf16x8 h[2]; };
__device__ __forceinline__ bf16x16 ld_frag(const unsigned short* p0,
                                           const unsigned short* p1) {
    FragU f;
    f.h[0] = *(const bf16x8*)p0;
    f.h[1] = *(const bf16x8*)p1;
    return f.v;
}

// ---------------- conversion kernels ----------------
__global__ __launch_bounds__(256) void cvt2bf(const float* __restrict__ in,
                                              unsigned short* __restrict__ out, int n) {
    int i = blockIdx.x * 256 + threadIdx.x;
    if (i < n) out[i] = f2bf(in[i]);
}

// convert with row zero-padding (for W_xproj 96 -> 128 rows)
__global__ __launch_bounds__(256) void cvt2bf_pad(const float* __restrict__ in,
                                                  unsigned short* __restrict__ out,
                                                  int rows_in, int cols, int n_out) {
    int i = blockIdx.x * 256 + threadIdx.x;
    if (i >= n_out) return;
    int r = i / cols;
    out[i] = (r < rows_in) ? f2bf(in[i]) : (unsigned short)0;
}

// extract dt columns (0..63) of dbc (ld 128) into bf16 (8192 x 64)
__global__ __launch_bounds__(256) void extract_dt(const float* __restrict__ dbc,
                                                  unsigned short* __restrict__ dt) {
    int i = blockIdx.x * 256 + threadIdx.x;   // NTOK*DTR threads
    int r = i >> 6, c = i & 63;
    dt[i] = f2bf(dbc[(size_t)r * DBCP + c]);
}

// ---------------- bf16 WMMA GEMM:  C(MxN) = A(MxK) * W(NxK)^T ----------------
// block: 256 threads = 8 waves; wave tile 32x64 -> block tile 256x64.
// grid = (N/64, M/256). A, W are bf16 K-major. K must be a multiple of 64.
// Ping-pong pipelined k-loop (unroll 2, two independent fragment buffers):
// no register-rotation copies, each buffer's reload issues right after its
// consuming WMMAs and overlaps the other buffer's 8-WMMA chain.
// act: 0 = none, 1 = softplus.
__global__ __launch_bounds__(256)
void wgemm_bf16(const unsigned short* __restrict__ A, int lda,
                const unsigned short* __restrict__ W, int ldb,
                const float* __restrict__ bias,
                float* __restrict__ Cf, unsigned short* __restrict__ Cb, int ldc,
                int K, int act) {
    const int lane = threadIdx.x & 31;
    const int wave = threadIdx.x >> 5;
    const int half = lane >> 4;
    const int l16  = lane & 15;
    const int m0   = (blockIdx.y * 8 + wave) * 32;
    const int n0   = blockIdx.x * 64;

    f32x8 acc[2][4];
#pragma unroll
    for (int i = 0; i < 2; ++i)
#pragma unroll
        for (int j = 0; j < 4; ++j)
            acc[i][j] = (f32x8){0.f, 0.f, 0.f, 0.f, 0.f, 0.f, 0.f, 0.f};

    // A fragment base: row m0+l16 (and +16), element offset half*8;
    // per k-step the two 16B chunks sit at +k0 and +k0+16.
    const unsigned short* arow0 = A + (size_t)(m0 + l16) * lda + half * 8;
    const unsigned short* arow1 = arow0 + (size_t)16 * lda;
    // B fragment base: row n (=column of C), element offset half*8;
    // per k-step the two 16B chunks sit at +k0 and +k0+16.
    const unsigned short* brow[4];
#pragma unroll
    for (int nt = 0; nt < 4; ++nt)
        brow[nt] = W + (size_t)(n0 + nt * 16 + l16) * ldb + half * 8;

    // two fragment buffers (ping-pong)
    bf16x16 a0A, a1A, bA[4];
    bf16x16 a0B, a1B, bB[4];

#define LOAD_STAGE(a0x, a1x, bx, kk)                                          \
    do {                                                                      \
        a0x = ld_frag(arow0 + (kk), arow0 + (kk) + 16);                       \
        a1x = ld_frag(arow1 + (kk), arow1 + (kk) + 16);                       \
        _Pragma("unroll")                                                     \
        for (int nt = 0; nt < 4; ++nt)                                        \
            bx[nt] = ld_frag(brow[nt] + (kk), brow[nt] + (kk) + 16);          \
    } while (0)

#define COMPUTE_STAGE(a0x, a1x, bx)                                           \
    do {                                                                      \
        _Pragma("unroll")                                                     \
        for (int nt = 0; nt < 4; ++nt) {                                      \
            acc[0][nt] = __builtin_amdgcn_wmma_f32_16x16x32_bf16(             \
                false, a0x, false, bx[nt], (short)0, acc[0][nt], false, false);\
            acc[1][nt] = __builtin_amdgcn_wmma_f32_16x16x32_bf16(             \
                false, a1x, false, bx[nt], (short)0, acc[1][nt], false, false);\
        }                                                                     \
    } while (0)

    LOAD_STAGE(a0A, a1A, bA, 0);
    LOAD_STAGE(a0B, a1B, bB, 32);

    for (int k0 = 64; k0 < K; k0 += 64) {
        COMPUTE_STAGE(a0A, a1A, bA);
        LOAD_STAGE(a0A, a1A, bA, k0);
        COMPUTE_STAGE(a0B, a1B, bB);
        LOAD_STAGE(a0B, a1B, bB, k0 + 32);
    }
    COMPUTE_STAGE(a0A, a1A, bA);
    COMPUTE_STAGE(a0B, a1B, bB);

#undef LOAD_STAGE
#undef COMPUTE_STAGE

#pragma unroll
    for (int mt = 0; mt < 2; ++mt) {
#pragma unroll
        for (int nt = 0; nt < 4; ++nt) {
            const int col = n0 + nt * 16 + l16;
            const float bv = bias ? bias[col] : 0.f;
#pragma unroll
            for (int r = 0; r < 8; ++r) {
                const int row = m0 + mt * 16 + half * 8 + r;
                float v = acc[mt][nt][r] + bv;
                if (act == 1) v = softplusf(v);
                const size_t idx = (size_t)row * ldc + col;
                if (Cf) Cf[idx] = v;
                if (Cb) Cb[idx] = f2bf(v);
            }
        }
    }
}

// ---------------- causal depthwise conv1d (k=4) + SiLU ----------------
// u = cols [0, DI) of xz (ld 2*DI); out u_bf (NTOK x DI)
__global__ __launch_bounds__(256)
void conv_silu(const unsigned short* __restrict__ xz,
               const float* __restrict__ cw, const float* __restrict__ cb,
               unsigned short* __restrict__ u_out) {
    int i = blockIdx.x * 256 + threadIdx.x;   // NB*LSEQ*DI threads
    int d = i & (DI - 1);
    int l = (i >> 11) & (LSEQ - 1);
    int b = i >> 22;
    float acc = cb[d];
#pragma unroll
    for (int k = 0; k < 4; ++k) {
        int ls = l - 3 + k;
        if (ls >= 0) {
            float uv = bf2f(xz[((size_t)(b * LSEQ + ls)) * (2 * DI) + d]);
            acc += uv * cw[d * 4 + k];
        }
    }
    u_out[i] = f2bf(siluf(acc));
}

// ---------------- selective scan (serial in L, parallel over B*DI channels) ---
// fuses: h = exp(dt*A)*h + (dt*u)*B ; y = h.C ; y = (y + u*D) * silu(z)
__global__ __launch_bounds__(256)
void sscan(const float* __restrict__ delta,      // NTOK x DI (f32)
           const unsigned short* __restrict__ u_bf,   // NTOK x DI
           const unsigned short* __restrict__ xz_bf,  // NTOK x 2*DI (z at +DI)
           const float* __restrict__ dbc,        // NTOK x DBCP (B at 64, C at 80)
           const float* __restrict__ Alog,       // DI x 16
           const float* __restrict__ Dp,         // DI
           unsigned short* __restrict__ y_bf) {  // NTOK x DI
    __shared__ float sBC[256];                   // 8 timesteps x 32 (B|C)
    const int tid = threadIdx.x;
    const int d = blockIdx.x * 256 + tid;
    const int b = blockIdx.y;
    const size_t row0 = (size_t)b * LSEQ;

    float Av[DSTATE], h[DSTATE];
#pragma unroll
    for (int s = 0; s < DSTATE; ++s) {
        Av[s] = -__expf(Alog[d * DSTATE + s]);
        h[s]  = 0.f;
    }
    const float Dv = Dp[d];

    for (int l0 = 0; l0 < LSEQ; l0 += 8) {
        __syncthreads();
        {   // cooperative stage of B_ssm|C_ssm for 8 timesteps
            int step = tid >> 5, j = tid & 31;
            sBC[tid] = dbc[(row0 + l0 + step) * DBCP + DTR + j];
        }
        __syncthreads();
#pragma unroll
        for (int t = 0; t < 8; ++t) {
            const size_t rid = row0 + l0 + t;
            const float dl = delta[rid * DI + d];
            const float uv = bf2f(u_bf[rid * DI + d]);
            const float zv = bf2f(xz_bf[rid * (2 * DI) + DI + d]);
            const float du = dl * uv;
            float y = 0.f;
#pragma unroll
            for (int s = 0; s < DSTATE; ++s) {
                float dA = __expf(dl * Av[s]);
                h[s] = dA * h[s] + du * sBC[t * 32 + s];
                y += h[s] * sBC[t * 32 + 16 + s];
            }
            y = (y + uv * Dv) * siluf(zv);
            y_bf[rid * DI + d] = f2bf(y);
        }
    }
}

// ---------------- launcher ----------------
extern "C" void kernel_launch(void* const* d_in, const int* in_sizes, int n_in,
                              void* d_out, int out_size, void* d_ws, size_t ws_size,
                              hipStream_t stream) {
    const float* x     = (const float*)d_in[0];
    const float* Wfc1  = (const float*)d_in[1];
    const float* bfc1  = (const float*)d_in[2];
    const float* Win   = (const float*)d_in[3];
    const float* convw = (const float*)d_in[4];
    const float* convb = (const float*)d_in[5];
    const float* Wxp   = (const float*)d_in[6];
    const float* Wdt   = (const float*)d_in[7];
    const float* bdt   = (const float*)d_in[8];
    const float* Alog  = (const float*)d_in[9];
    const float* Dp    = (const float*)d_in[10];
    const float* Wout  = (const float*)d_in[11];
    const float* Wfc2  = (const float*)d_in[12];
    const float* bfc2  = (const float*)d_in[13];

    char* ws = (char*)d_ws;
    size_t off = 0;
    auto alloc = [&](size_t bytes) -> void* {
        void* p = ws + off;
        off = (off + bytes + 255) & ~(size_t)255;
        return p;
    };
    unsigned short* x_bf    = (unsigned short*)alloc((size_t)NTOK * DIN * 2);
    unsigned short* wfc1_bf = (unsigned short*)alloc((size_t)DM * DIN * 2);
    unsigned short* win_bf  = (unsigned short*)alloc((size_t)2 * DI * DM * 2);
    unsigned short* wxp_bf  = (unsigned short*)alloc((size_t)DBCP * DI * 2);
    unsigned short* wdt_bf  = (unsigned short*)alloc((size_t)DI * DTR * 2);
    unsigned short* wout_bf = (unsigned short*)alloc((size_t)DM * DI * 2);
    unsigned short* wfc2_bf = (unsigned short*)alloc((size_t)DOUTW * DM * 2);
    unsigned short* h1_bf   = (unsigned short*)alloc((size_t)NTOK * DM * 2);
    unsigned short* xz_bf   = (unsigned short*)alloc((size_t)NTOK * 2 * DI * 2);
    unsigned short* u_bf    = (unsigned short*)alloc((size_t)NTOK * DI * 2);
    float*          dbc     = (float*)alloc((size_t)NTOK * DBCP * 4);
    unsigned short* dt_bf   = (unsigned short*)alloc((size_t)NTOK * DTR * 2);
    float*          delta   = (float*)alloc((size_t)NTOK * DI * 4);
    unsigned short* y_bf    = (unsigned short*)alloc((size_t)NTOK * DI * 2);
    unsigned short* m_bf    = (unsigned short*)alloc((size_t)NTOK * DM * 2);

    auto cvt = [&](const float* in, unsigned short* out, int n) {
        cvt2bf<<<dim3((n + 255) / 256), dim3(256), 0, stream>>>(in, out, n);
    };
    // weight + input conversion to bf16
    cvt(x, x_bf, NTOK * DIN);
    cvt(Wfc1, wfc1_bf, DM * DIN);
    cvt(Win, win_bf, 2 * DI * DM);
    cvt(Wdt, wdt_bf, DI * DTR);
    cvt(Wout, wout_bf, DM * DI);
    cvt(Wfc2, wfc2_bf, DOUTW * DM);
    {
        int n = DBCP * DI;
        cvt2bf_pad<<<dim3((n + 255) / 256), dim3(256), 0, stream>>>(
            Wxp, wxp_bf, DBCW, DI, n);
    }

    const dim3 blk(256);
    const int MB = NTOK / 256;   // 32 block rows (256 rows per block)

    // fc1: h1 = x @ Wfc1^T + b   (bf16 out)
    wgemm_bf16<<<dim3(DM / 64, MB), blk, 0, stream>>>(
        x_bf, DIN, wfc1_bf, DIN, bfc1, nullptr, h1_bf, DM, DIN, 0);

    // in-proj: xz = h1 @ Win^T   (bf16 out, 8192 x 4096)
    wgemm_bf16<<<dim3(2 * DI / 64, MB), blk, 0, stream>>>(
        h1_bf, DM, win_bf, DM, nullptr, nullptr, xz_bf, 2 * DI, DM, 0);

    // depthwise causal conv + SiLU
    conv_silu<<<dim3((NB * LSEQ * DI) / 256), blk, 0, stream>>>(
        xz_bf, convw, convb, u_bf);

    // x-proj: dbc = u @ Wxp^T    (f32 out, padded N=128)
    wgemm_bf16<<<dim3(DBCP / 64, MB), blk, 0, stream>>>(
        u_bf, DI, wxp_bf, DI, nullptr, dbc, nullptr, DBCP, DI, 0);

    // dt extraction -> bf16
    extract_dt<<<dim3((NTOK * DTR) / 256), blk, 0, stream>>>(dbc, dt_bf);

    // delta = softplus(dt @ Wdt^T + b_dt)  (f32 out)
    wgemm_bf16<<<dim3(DI / 64, MB), blk, 0, stream>>>(
        dt_bf, DTR, wdt_bf, DTR, bdt, delta, nullptr, DI, DTR, 1);

    // selective scan + D skip + SiLU(z) gating (bf16 out)
    sscan<<<dim3(DI / 256, NB), blk, 0, stream>>>(
        delta, u_bf, xz_bf, dbc, Alog, Dp, y_bf);

    // out-proj: m = y @ Wout^T   (bf16 out)
    wgemm_bf16<<<dim3(DM / 64, MB), blk, 0, stream>>>(
        y_bf, DI, wout_bf, DI, nullptr, nullptr, m_bf, DM, DI, 0);

    // fc2: out = m @ Wfc2^T + b  (f32 out -> d_out)
    wgemm_bf16<<<dim3(DOUTW / 64, MB), blk, 0, stream>>>(
        m_bf, DM, wfc2_bf, DM, bfc2, (float*)d_out, nullptr, DOUTW, DM, 0);
}
